// CrossAttention_51634096832973
// MI455X (gfx1250) — matrix operand order repaired and checked by reading, not compiled, not run on hardware
//
#include <hip/hip_runtime.h>

// ---------- CDNA5 WMMA types ----------
typedef __attribute__((ext_vector_type(16))) __bf16 v16bf;
typedef __attribute__((ext_vector_type(8)))  float  v8f;

union FragBF16 {
    v16bf v;
    __bf16 e[16];
    uint4  q[2];
};

// ---------- load 8 elements -> 8 bf16 ----------
__device__ __forceinline__ void load8_bf16(const float* __restrict__ p, __bf16* o) {
    const float4* p4 = reinterpret_cast<const float4*>(p);
    float4 a = p4[0];
    float4 b = p4[1];
    o[0] = (__bf16)a.x; o[1] = (__bf16)a.y; o[2] = (__bf16)a.z; o[3] = (__bf16)a.w;
    o[4] = (__bf16)b.x; o[5] = (__bf16)b.y; o[6] = (__bf16)b.z; o[7] = (__bf16)b.w;
}
__device__ __forceinline__ void load8_bf16(const __bf16* __restrict__ p, __bf16* o) {
    *reinterpret_cast<uint4*>(o) = *reinterpret_cast<const uint4*>(p);
}

__device__ __forceinline__ void store_c(float* p, float v)  { *p = v; }
__device__ __forceinline__ void store_c(__bf16* p, float v) { *p = (__bf16)v; }

// =====================================================================
// Tiled GEMM:  C[M,N] = A[M,K] * B   (bf16 math, f32 accumulate, WMMA)
//   B_NK = true : B stored (N,K) row-major  ->  C = A * B^T
//   B_NK = false: B stored (K,N) row-major  ->  C = A * B
// Block: 256 threads (8 wave32). Block tile: 256(M) x 64(N). K-step: 32.
// Each wave: 32(M) x 64(N) = 8 x v_wmma_f32_16x16x32_bf16 per K-step.
// B tile staged in LDS as [n][k] bf16 (stride 40). When B is already bf16
// and (N,K)-major, staging uses global_load_async_to_lds_b128 (ASYNCcnt).
// =====================================================================
#define LDS_STRIDE 40

template <typename TA, typename TB, typename TC, bool B_NK>
__global__ __launch_bounds__(256) void gemm_wmma_bf16(
    const TA* __restrict__ A, long long sA, int lda,
    const TB* __restrict__ B, long long sB, int ldb,
    TC* __restrict__ C, long long sC, int ldc,
    int K)
{
    __shared__ __bf16 lsB[64 * LDS_STRIDE];   // only LDS object -> base offset 0

    const int tid  = threadIdx.x;
    const int lane = tid & 31;
    const int wave = tid >> 5;

    const long long zb = blockIdx.z;
    const TA* Ab = A + zb * sA;
    const TB* Bb = B + zb * sB;
    TC*       Cb = C + zb * sC;

    const int M0 = blockIdx.y * 256;
    const int N0 = blockIdx.x * 64;

    v8f acc[2][4];
#pragma unroll
    for (int s = 0; s < 2; ++s)
#pragma unroll
        for (int t = 0; t < 4; ++t)
#pragma unroll
            for (int r = 0; r < 8; ++r) acc[s][t][r] = 0.0f;

    // Per-lane A-fragment addressing (ISA 16-bit A layout):
    //   lanes 0-15 : M=lane,    K = {0..7} U {16..23}
    //   lanes 16-31: M=lane-16, K = {8..15} U {24..31}
    const int am0   = M0 + wave * 32 + (lane & 15);   // first 16-row subtile
    const int am1   = am0 + 16;                        // second 16-row subtile
    const int akoff = (lane >> 4) * 8;

    // B staging thread mapping (B_NK): n = tid/4 in [0,64), 8 consecutive k
    const int bn    = tid >> 2;
    const int bkOff = (tid & 3) * 8;
    // B staging thread mapping (K,N): k = tid/8 in [0,32), 8 consecutive n
    const int bk    = tid >> 3;
    const int bnOff = (tid & 7) * 8;

    for (int k0 = 0; k0 < K; k0 += 32) {
        // ---- stage B tile (64 N x 32 K) into LDS as [n][k] bf16 ----
        if constexpr (B_NK) {
            if constexpr (__is_same(TB, __bf16)) {
                // bf16, K-contiguous: pure 16B/lane copy -> async LDS DMA
                const __bf16* gp = Bb + (long long)(N0 + bn) * ldb + (k0 + bkOff);
                unsigned long long ga = (unsigned long long)gp;
                unsigned ldsoff = (unsigned)((bn * LDS_STRIDE + bkOff) * 2);
                asm volatile("global_load_async_to_lds_b128 %0, %1, off"
                             :: "v"(ldsoff), "v"(ga) : "memory");
                asm volatile("s_wait_asynccnt 0x0" ::: "memory");
            } else {
                __bf16 tmp[8];
                load8_bf16(Bb + (long long)(N0 + bn) * ldb + (k0 + bkOff), tmp);
                *reinterpret_cast<uint4*>(&lsB[bn * LDS_STRIDE + bkOff]) =
                    *reinterpret_cast<uint4*>(tmp);
            }
        } else {
            // B is (K,N) row-major: transpose into LDS
            __bf16 tmp[8];
            load8_bf16(Bb + (long long)(k0 + bk) * ldb + (N0 + bnOff), tmp);
#pragma unroll
            for (int i = 0; i < 8; ++i)
                lsB[(bnOff + i) * LDS_STRIDE + bk] = tmp[i];
        }
        __syncthreads();

        // ---- A fragments straight from global (bf16 in regs) ----
        FragBF16 fa0, fa1;
        load8_bf16(Ab + (long long)am0 * lda + (k0 + akoff),      &fa0.e[0]);
        load8_bf16(Ab + (long long)am0 * lda + (k0 + akoff + 16), &fa0.e[8]);
        load8_bf16(Ab + (long long)am1 * lda + (k0 + akoff),      &fa1.e[0]);
        load8_bf16(Ab + (long long)am1 * lda + (k0 + akoff + 16), &fa1.e[8]);

        // prefetch next K-step of A
        if (k0 + 32 < K) {
            __builtin_prefetch(Ab + (long long)am0 * lda + (k0 + 32 + akoff), 0, 1);
            __builtin_prefetch(Ab + (long long)am1 * lda + (k0 + 32 + akoff), 0, 1);
        }

        // ---- all 4 B fragments first (clause the ds_loads), then 8 WMMAs ----
        const int nl = lane & 15;
        const int kl = (lane >> 4) * 8;
        FragBF16 fb[4];
#pragma unroll
        for (int t = 0; t < 4; ++t) {
            const uint4* src = reinterpret_cast<const uint4*>(
                &lsB[(t * 16 + nl) * LDS_STRIDE + kl]);
            fb[t].q[0] = src[0];
            fb[t].q[1] = src[2];
        }
#pragma unroll
        for (int t = 0; t < 4; ++t) {
            acc[0][t] = __builtin_amdgcn_wmma_f32_16x16x32_bf16(
                false, fa0.v, false, fb[t].v, (short)0, acc[0][t], false, false);
            acc[1][t] = __builtin_amdgcn_wmma_f32_16x16x32_bf16(
                false, fa1.v, false, fb[t].v, (short)0, acc[1][t], false, false);
        }
        __syncthreads();
    }

    // ---- store C (32-bit C/D layout: VGPR r -> M=r / M=8+r by lane half) ----
    const int cn = N0 + (lane & 15);
#pragma unroll
    for (int s = 0; s < 2; ++s) {
        const int cm = M0 + wave * 32 + s * 16 + ((lane >> 4) << 3);
#pragma unroll
        for (int t = 0; t < 4; ++t)
#pragma unroll
            for (int r = 0; r < 8; ++r)
                store_c(&Cb[(long long)(cm + r) * ldc + (cn + t * 16)], acc[s][t][r]);
    }
}

// =====================================================================
// In-place row softmax over bf16 logits: 1024 cols, 256 threads x 4 cols.
// =====================================================================
__global__ __launch_bounds__(256) void softmax_rows_bf16(__bf16* __restrict__ S, float scale)
{
    __shared__ float red[256];
    const int tid = threadIdx.x;
    __bf16* row = S + (long long)blockIdx.x * 1024;

    union { uint2 u; __bf16 e[4]; } pkt;
    pkt.u = *reinterpret_cast<uint2*>(row + tid * 4);

    float f[4];
    float lmax = -3.0e38f;
#pragma unroll
    for (int i = 0; i < 4; ++i) {
        f[i] = (float)pkt.e[i] * scale;
        lmax = fmaxf(lmax, f[i]);
    }
    red[tid] = lmax;
    __syncthreads();
    for (int s = 128; s > 0; s >>= 1) {
        if (tid < s) red[tid] = fmaxf(red[tid], red[tid + s]);
        __syncthreads();
    }
    const float m = red[0];
    __syncthreads();

    float lsum = 0.0f;
#pragma unroll
    for (int i = 0; i < 4; ++i) {
        f[i] = __expf(f[i] - m);
        lsum += f[i];
    }
    red[tid] = lsum;
    __syncthreads();
    for (int s = 128; s > 0; s >>= 1) {
        if (tid < s) red[tid] += red[tid + s];
        __syncthreads();
    }
    const float inv = 1.0f / red[0];

#pragma unroll
    for (int i = 0; i < 4; ++i) pkt.e[i] = (__bf16)(f[i] * inv);
    *reinterpret_cast<uint2*>(row + tid * 4) = pkt.u;
}

// =====================================================================
// Launcher
//   b=16, c=1024, dim=1024
//   ws: q bf16 (16384x1024) | kv bf16 (16384x2048) | S bf16 (16x1024x1024)
// =====================================================================
extern "C" void kernel_launch(void* const* d_in, const int* in_sizes, int n_in,
                              void* d_out, int out_size, void* d_ws, size_t ws_size,
                              hipStream_t stream)
{
    const float*  x    = (const float*)d_in[0];   // (16,1024,32,32) -> (16384,1024)
    const float*  cond = (const float*)d_in[1];
    const float*  Wq   = (const float*)d_in[2];   // (1024,1024)
    const float*  Wkv  = (const float*)d_in[3];   // (2048,1024)
    float*        out  = (float*)d_out;           // (16,1024,1024)

    char* ws = (char*)d_ws;
    __bf16* qb  = (__bf16*)(ws);                          // 33,554,432 B
    __bf16* kvb = (__bf16*)(ws + 33554432);               // 67,108,864 B
    __bf16* Sb  = (__bf16*)(ws + 33554432 + 67108864);    // 33,554,432 B
    (void)in_sizes; (void)n_in; (void)out_size; (void)ws_size;

    dim3 blk(256);

    // q = x @ Wq^T   (M=16384, N=1024, K=1024)
    gemm_wmma_bf16<float, float, __bf16, true>
        <<<dim3(1024 / 64, 16384 / 256, 1), blk, 0, stream>>>(
            x, 0LL, 1024, Wq, 0LL, 1024, qb, 0LL, 1024, 1024);

    // kv = cond @ Wkv^T  (M=16384, N=2048, K=1024); k = cols[0:1024), v = cols[1024:2048)
    gemm_wmma_bf16<float, float, __bf16, true>
        <<<dim3(2048 / 64, 16384 / 256, 1), blk, 0, stream>>>(
            cond, 0LL, 1024, Wkv, 0LL, 1024, kvb, 0LL, 2048, 1024);

    // S = q @ k^T per batch  (M=1024, N=1024, K=1024, 16 batches) -- async-LDS B staging
    gemm_wmma_bf16<__bf16, __bf16, __bf16, true>
        <<<dim3(1024 / 64, 1024 / 256, 16), blk, 0, stream>>>(
            qb, 1048576LL, 1024, kvb, 2097152LL, 2048, Sb, 1048576LL, 1024, 1024);

    // softmax(scale * S) row-wise, in place (scale = 1024^-0.5 = 1/32)
    softmax_rows_bf16<<<dim3(16384), blk, 0, stream>>>(Sb, 0.03125f);

    // out = P @ v per batch  (B is (K,N) = v rows, fp32 output to d_out)
    gemm_wmma_bf16<__bf16, __bf16, float, false>
        <<<dim3(1024 / 64, 1024 / 256, 16), blk, 0, stream>>>(
            Sb, 1048576LL, 1024, kvb + 1024, 2097152LL, 2048, out, 1048576LL, 1024, 1024);
}
